// SecurityGraphEncoder_78073915507130
// MI455X (gfx1250) — compile-verified, matrix operand order checked by reading, not run on hardware
//
#include <hip/hip_runtime.h>
#include <hip/hip_bf16.h>

#define Bq 2
#define Nn 1536
#define NFD 128
#define Hd 256
#define NHEADS 8
#define HDIM 32
#define ETYPES 10
#define RSF 40   // LDS tile row stride in halves (32 data + 8 pad)

typedef __attribute__((ext_vector_type(16))) _Float16 v16h;
typedef __attribute__((ext_vector_type(8)))  float    v8f;

__device__ __forceinline__ v8f wmma16(v16h a, v16h b, v8f c) {
  return __builtin_amdgcn_wmma_f32_16x16x32_f16(false, a, false, b, (short)0, c, false, false);
}

// Load a 16x32 f16 fragment (A layout; B uses the symmetric lane=N convention).
// Lane l: idx=l&15 selects row; K halves split at 8/24. rstride in halves (even).
__device__ __forceinline__ v16h ld_frag(const _Float16* base, int rstride) {
  int l = threadIdx.x & 31;
  int idx = l & 15;
  int kh = (l >> 4) << 3;             // 0 or 8
  const _Float16* p = base + (size_t)idx * rstride + kh;
  union { v16h v; unsigned u[8]; } f;
#pragma unroll
  for (int r = 0; r < 4; ++r) f.u[r]     = *(const unsigned*)(p + 2 * r);
#pragma unroll
  for (int r = 0; r < 4; ++r) f.u[4 + r] = *(const unsigned*)(p + 16 + 2 * r);
  return f.v;
}

// ---------------------------------------------------------------------------
// GEMM: C[M x Nout] = A[M x K](f32->f16) @ W[K x Nout] + bias
// Block tile 128x64, 8 waves each 32x32 (2x2 WMMA). K % 32 == 0, M = 3072.
// WF32: write f32 C. C16M: 0 none, 1 row-major f16 mirror (ldc16),
//       2 head-transposed f16 mirror vt16[(b*8+h)*32+d][j] (Nout<=256).
// ---------------------------------------------------------------------------
template<int WF32, int C16M, int RESID, int RELU>
__global__ __launch_bounds__(256)
void gemm_wmma(const float* __restrict__ A, int lda,
               const float* __restrict__ W, int ldw,
               const float* __restrict__ bias,
               const float* __restrict__ resid,
               float* __restrict__ C, int ldc,
               _Float16* __restrict__ C16, int ldc16,
               int K)
{
  __shared__ _Float16 As[128 * RSF];
  __shared__ _Float16 Wt[64 * RSF];
  const int m0 = blockIdx.y * 128;
  const int n0 = blockIdx.x * 64;
  const int t = threadIdx.x;
  const int lane = t & 31;
  const int wave = t >> 5;
  const int wy = wave & 3, wx = wave >> 2;
  v8f acc[2][2] = {};
  for (int k0 = 0; k0 < K; k0 += 32) {
    __syncthreads();
#pragma unroll
    for (int e = 0; e < 8; ++e) {               // A tile 128x32 -> f16 (packed b32 stores)
      int idx = t + 256 * e;                    // 0..2047, pair id
      int row = idx >> 4, kp = (idx & 15) << 1;
      const float* ap = &A[(size_t)(m0 + row) * lda + k0 + kp];
      union { unsigned u; _Float16 h[2]; } pk;
      pk.h[0] = (_Float16)ap[0];
      pk.h[1] = (_Float16)ap[1];
      *(unsigned*)&As[row * RSF + kp] = pk.u;
    }
#pragma unroll
    for (int e = 0; e < 4; ++e) {               // W tile 32x64 -> transposed f16
      int idx = t + 256 * e;                    // 0..1023
      int n = idx & 63, kp = (idx >> 6) << 1;
      const float* wp = &W[(size_t)(k0 + kp) * ldw + n0 + n];
      union { unsigned u; _Float16 h[2]; } pk;
      pk.h[0] = (_Float16)wp[0];
      pk.h[1] = (_Float16)wp[ldw];
      *(unsigned*)&Wt[n * RSF + kp] = pk.u;
    }
    // prefetch next K-step tiles
    __builtin_prefetch(&A[(size_t)(m0 + (t >> 1)) * lda + k0 + 32], 0, 1);
    __builtin_prefetch(&W[(size_t)(k0 + 32 + (t >> 6)) * ldw + n0 + (t & 63)], 0, 1);
    __syncthreads();
    v16h a0 = ld_frag(As + (wy * 32) * RSF, RSF);
    v16h a1 = ld_frag(As + (wy * 32 + 16) * RSF, RSF);
    v16h b0 = ld_frag(Wt + (wx * 32) * RSF, RSF);
    v16h b1 = ld_frag(Wt + (wx * 32 + 16) * RSF, RSF);
    acc[0][0] = wmma16(a0, b0, acc[0][0]);
    acc[0][1] = wmma16(a0, b1, acc[0][1]);
    acc[1][0] = wmma16(a1, b0, acc[1][0]);
    acc[1][1] = wmma16(a1, b1, acc[1][1]);
  }
  const int half = lane >> 4;
  const int cidx = lane & 15;
#pragma unroll
  for (int mi = 0; mi < 2; ++mi)
#pragma unroll
    for (int ni = 0; ni < 2; ++ni) {
      int colg = n0 + wx * 32 + ni * 16 + cidx;
      float bv = bias[colg];
#pragma unroll
      for (int e = 0; e < 8; ++e) {
        int rowg = m0 + wy * 32 + mi * 16 + e + 8 * half;
        float v = acc[mi][ni][e] + bv;
        if (RESID) v += resid[(size_t)rowg * ldc + colg];
        if (RELU) v = fmaxf(v, 0.f);
        if (WF32) C[(size_t)rowg * ldc + colg] = v;
        if (C16M == 1) C16[(size_t)rowg * ldc16 + colg] = (_Float16)v;
        if (C16M == 2) {
          int bb = (rowg >= Nn) ? 1 : 0;
          int jj = rowg - bb * Nn;
          C16[((size_t)(bb * 256 + colg)) * Nn + jj] = (_Float16)v;
        }
      }
    }
}

// Pack adjacency union + edge types into one byte: 0xFF = masked, else edge type.
__global__ __launch_bounds__(256)
void prep_mask(const int* __restrict__ etm, const int* __restrict__ a1,
               const int* __restrict__ a2, const int* __restrict__ a3,
               unsigned char* __restrict__ mb)
{
  int i = blockIdx.x * 256 + threadIdx.x;
  if (i >= Bq * Nn * Nn) return;
  int adj = a1[i] | a2[i] | a3[i];
  mb[i] = adj ? (unsigned char)etm[i] : (unsigned char)0xFF;
}

// x0[:, 256:320] = type_emb[node_types]
__global__ __launch_bounds__(256)
void type_gather(const int* __restrict__ nt, const float* __restrict__ te,
                 float* __restrict__ x0)
{
  int idx = blockIdx.x * 256 + threadIdx.x;
  if (idx >= Bq * Nn * 64) return;
  int d = idx & 63, n = idx >> 6;
  x0[(size_t)n * 320 + 256 + d] = te[nt[n] * 64 + d];
}

// Per-row score terms: sq = q.a1, sk = k.a2, p[e] = q.ete[e]
__global__ __launch_bounds__(256)
void gat_pre(const float* __restrict__ q, const float* __restrict__ k,
             const float* __restrict__ att, const float* __restrict__ ete,
             float* __restrict__ sq, float* __restrict__ sk, float* __restrict__ p)
{
  int idx = blockIdx.x * 256 + threadIdx.x;
  if (idx >= Bq * Nn * NHEADS * 12) return;
  int task = idx % 12;
  int bnh = idx / 12;
  int h = bnh % NHEADS;
  int bn = bnh / NHEADS;
  int b = bn / Nn, n = bn % Nn;
  size_t qoff = (size_t)bn * Hd + h * HDIM;
  float s = 0.f;
  if (task == 0) {
    for (int d = 0; d < HDIM; ++d) s += q[qoff + d] * att[h * 64 + d];
    sq[((size_t)(b * NHEADS + h)) * Nn + n] = s;
  } else if (task == 1) {
    for (int d = 0; d < HDIM; ++d) s += k[qoff + d] * att[h * 64 + 32 + d];
    sk[((size_t)(b * NHEADS + h)) * Nn + n] = s;
  } else {
    int e = task - 2;
    for (int d = 0; d < HDIM; ++d) s += q[qoff + d] * ete[e * HDIM + d];
    p[(((size_t)(b * NHEADS + h)) * Nn + n) * ETYPES + e] = s;
  }
}

// Compute one score in leaky-relu'd, masked form. c==0xFF -> masked.
__device__ __forceinline__ float gat_score(unsigned c, float msq, float skj,
                                           const float* __restrict__ prow)
{
  unsigned cc = c < 9u ? c : 9u;          // clamp so the gather is always in-bounds
  float s = msq + skj + prow[cc];
  s = (s > 0.f) ? s : 0.2f * s;
  return (c == 0xFFu) ? -1.0e30f : s;
}

// Fused GAT attention, barrier-free: streaming 2-pass softmax with the P tile
// built directly in the WMMA A-fragment register layout; V read from the
// head-transposed f16 mirror vt16[(b*8+h)*32+d][j]. 4 waves/block, 16 rows each.
__global__ __launch_bounds__(128)
void gat_attn(const float* __restrict__ sq, const float* __restrict__ sk,
              const float* __restrict__ p, const unsigned char* __restrict__ mb,
              const _Float16* __restrict__ vt16, float* __restrict__ out)
{
  int wave = threadIdx.x >> 5, lane = threadIdx.x & 31;
  int tile = blockIdx.x * 4 + wave;
  int i16 = tile % (Nn / 16);
  int bh = tile / (Nn / 16);
  int h = bh % NHEADS, b = bh / NHEADS;
  int i0 = i16 * 16;
  int r = lane & 15, jh = lane >> 4;
  int kh8 = 8 * jh;
  int irow = i0 + r;
  float msq = sq[(size_t)bh * Nn + irow];
  const float* prow = p + ((size_t)bh * Nn + irow) * ETYPES;
  const unsigned char* mrow = mb + ((size_t)b * Nn + irow) * Nn;
  const float* skrow = sk + (size_t)bh * Nn;

  // pass 1: row max / sum; row r is split across lanes r and r+16
  float m = -3.0e38f, sum = 0.f;
  for (int j0 = 0; j0 < Nn; j0 += 32) {
    unsigned long long w0 = *(const unsigned long long*)(mrow + j0 + kh8);
    unsigned long long w1 = *(const unsigned long long*)(mrow + j0 + 16 + kh8);
    float sv[16];
    float cm = -3.0e38f;
#pragma unroll
    for (int t = 0; t < 16; ++t) {
      int jloc = (t < 8) ? (kh8 + t) : (16 + kh8 + (t - 8));
      unsigned c = (unsigned)((((t < 8) ? w0 : w1) >> (8 * (t & 7))) & 0xFFu);
      float s = gat_score(c, msq, skrow[j0 + jloc], prow);
      sv[t] = s; cm = fmaxf(cm, s);
    }
    float nmx = fmaxf(m, cm);
    float a = 0.f;
#pragma unroll
    for (int t = 0; t < 16; ++t) a += __expf(sv[t] - nmx);
    sum = sum * __expf(m - nmx) + a;
    m = nmx;
  }
  {
    float om = __shfl_xor(m, 16);
    float osum = __shfl_xor(sum, 16);
    float nmx = fmaxf(m, om);
    sum = sum * __expf(m - nmx) + osum * __expf(om - nmx);
    m = nmx;
  }
  float inv = 1.f / sum;

  // pass 2: P fragment built in registers (element e <-> K = kh8+e / 16+kh8+e-8)
  const _Float16* vbase = vt16 + (size_t)bh * 32 * Nn;
  v8f acc0 = {}, acc1 = {};
  for (int j0 = 0; j0 < Nn; j0 += 32) {
    unsigned long long w0 = *(const unsigned long long*)(mrow + j0 + kh8);
    unsigned long long w1 = *(const unsigned long long*)(mrow + j0 + 16 + kh8);
    v16h pa;
#pragma unroll
    for (int t = 0; t < 16; ++t) {
      int jloc = (t < 8) ? (kh8 + t) : (16 + kh8 + (t - 8));
      unsigned c = (unsigned)((((t < 8) ? w0 : w1) >> (8 * (t & 7))) & 0xFFu);
      float s = gat_score(c, msq, skrow[j0 + jloc], prow);
      pa[t] = (_Float16)(__expf(s - m) * inv);
    }
    v16h b0 = ld_frag(vbase + j0, Nn);
    v16h b1 = ld_frag(vbase + (size_t)16 * Nn + j0, Nn);
    acc0 = wmma16(pa, b0, acc0);
    acc1 = wmma16(pa, b1, acc1);
  }
  int cidx = lane & 15, hb = 8 * (lane >> 4);
#pragma unroll
  for (int e = 0; e < 8; ++e) {
    float* orow = out + ((size_t)(b * Nn + i0 + e + hb)) * Hd + h * HDIM;
    orow[cidx] = acc0[e];
    orow[16 + cidx] = acc1[e];
  }
}

// MHA: QK^T via WMMA with fragments straight from global f16 (qk16, ld=512),
// streaming softmax, P staged via LDS (C-layout -> A-layout), V from vt16.
__global__ __launch_bounds__(128)
void mha_attn(const _Float16* __restrict__ qk16, const _Float16* __restrict__ vt16,
              float* __restrict__ out)
{
  __shared__ _Float16 Pt[4][16 * RSF];
  int wave = threadIdx.x >> 5, lane = threadIdx.x & 31;
  int tile = blockIdx.x * 4 + wave;
  int i16 = tile % (Nn / 16);
  int bh = tile / (Nn / 16);
  int h = bh % NHEADS, b = bh / NHEADS;
  int i0 = i16 * 16;
  const _Float16* qb = qk16 + ((size_t)(b * Nn + i0)) * 512 + h * HDIM;
  const _Float16* kb = qk16 + ((size_t)b * Nn) * 512 + 256 + h * HDIM;
  const _Float16* vbase = vt16 + (size_t)bh * 32 * Nn;
  v16h aq = ld_frag(qb, 512);
  const float scale = 0.17677669529663687f;   // 1/sqrt(32)
  float rm[8], rs[8];
#pragma unroll
  for (int e = 0; e < 8; ++e) { rm[e] = -3.0e38f; rs[e] = 0.f; }
  for (int j0 = 0; j0 < Nn; j0 += 16) {
    v16h bk = ld_frag(kb + (size_t)j0 * 512, 512);
    v8f s = {};
    s = wmma16(aq, bk, s);
#pragma unroll
    for (int e = 0; e < 8; ++e) {
      float sv = s[e] * scale;
      float cm = sv;
#pragma unroll
      for (int off = 1; off < 16; off <<= 1) cm = fmaxf(cm, __shfl_xor(cm, off));
      float nmx = fmaxf(rm[e], cm);
      float ex = __expf(sv - nmx);
#pragma unroll
      for (int off = 1; off < 16; off <<= 1) ex += __shfl_xor(ex, off);
      rs[e] = rs[e] * __expf(rm[e] - nmx) + ex;
      rm[e] = nmx;
    }
  }
  float rinv[8];
#pragma unroll
  for (int e = 0; e < 8; ++e) rinv[e] = 1.f / rs[e];
  _Float16* P = Pt[wave];
  v8f acc0 = {}, acc1 = {};
  int hb = 8 * (lane >> 4), cidx = lane & 15;
  for (int j0 = 0; j0 < Nn; j0 += 32) {
    __syncthreads();
#pragma unroll
    for (int jj0 = 0; jj0 < 32; jj0 += 16) {
      v16h bk = ld_frag(kb + (size_t)(j0 + jj0) * 512, 512);
      v8f s = {};
      s = wmma16(aq, bk, s);
#pragma unroll
      for (int e = 0; e < 8; ++e) {
        float w = __expf(s[e] * scale - rm[e]) * rinv[e];
        P[(e + hb) * RSF + jj0 + cidx] = (_Float16)w;
      }
    }
    __syncthreads();
    v16h af = ld_frag(P, RSF);
    acc0 = wmma16(af, ld_frag(vbase + j0, Nn), acc0);
    acc1 = wmma16(af, ld_frag(vbase + (size_t)16 * Nn + j0, Nn), acc1);
  }
#pragma unroll
  for (int e = 0; e < 8; ++e) {
    float* orow = out + ((size_t)(b * Nn + i0 + e + hb)) * Hd + h * HDIM;
    orow[cidx] = acc0[e];
    orow[16 + cidx] = acc1[e];
  }
}

__global__ __launch_bounds__(256)
void layernorm_k(const float* __restrict__ X, const float* __restrict__ g,
                 const float* __restrict__ bta, float* __restrict__ Y)
{
  __shared__ float red[8];
  int row = blockIdx.x, c = threadIdx.x;
  int lane = c & 31, wave = c >> 5;
  float v = X[(size_t)row * Hd + c];
  float s = v;
#pragma unroll
  for (int off = 1; off < 32; off <<= 1) s += __shfl_xor(s, off);
  if (lane == 0) red[wave] = s;
  __syncthreads();
  float tot = 0.f;
#pragma unroll
  for (int w = 0; w < 8; ++w) tot += red[w];
  float mean = tot * (1.f / 256.f);
  __syncthreads();
  float d = v - mean;
  float s2 = d * d;
#pragma unroll
  for (int off = 1; off < 32; off <<= 1) s2 += __shfl_xor(s2, off);
  if (lane == 0) red[wave] = s2;
  __syncthreads();
  float vt = 0.f;
#pragma unroll
  for (int w = 0; w < 8; ++w) vt += red[w];
  float var = vt * (1.f / 256.f);
  Y[(size_t)row * Hd + c] = d * rsqrtf(var + 1e-5f) * g[c] + bta[c];
}

__global__ __launch_bounds__(256)
void pool_mlp(const float* __restrict__ gr, const float* __restrict__ w1,
              const float* __restrict__ b1, const float* __restrict__ w2,
              const float* __restrict__ b2, float* __restrict__ outp)
{
  __shared__ float pooled[256];
  __shared__ float h1[256];
  int b = blockIdx.x, c = threadIdx.x;
  float s = 0.f;
  for (int n = 0; n < Nn; ++n) s += gr[((size_t)(b * Nn + n)) * Hd + c];
  pooled[c] = s * (1.f / (float)Nn);
  __syncthreads();
  float a = b1[c];
  for (int k = 0; k < 256; ++k) a += pooled[k] * w1[k * 256 + c];
  __syncthreads();
  h1[c] = fmaxf(a, 0.f);
  __syncthreads();
  float o = b2[c];
  for (int k = 0; k < 256; ++k) o += h1[k] * w2[k * 256 + c];
  outp[b * 256 + c] = o;
}

extern "C" void kernel_launch(void* const* d_in, const int* in_sizes, int n_in,
                              void* d_out, int out_size, void* d_ws, size_t ws_size,
                              hipStream_t stream) {
  (void)in_sizes; (void)n_in; (void)out_size; (void)ws_size;
  const float* node_features = (const float*)d_in[0];
  const int*   node_types    = (const int*)d_in[1];
  const int*   etm           = (const int*)d_in[2];
  const int*   adj1          = (const int*)d_in[3];
  const int*   adj2          = (const int*)d_in[4];
  const int*   adj3          = (const int*)d_in[5];
  const float* emb_w         = (const float*)d_in[6];
  const float* emb_b         = (const float*)d_in[7];
  const float* type_emb      = (const float*)d_in[8];
  const float* mha_in_w      = (const float*)d_in[45];
  const float* mha_in_b      = (const float*)d_in[46];
  const float* mha_out_w     = (const float*)d_in[47];
  const float* mha_out_b     = (const float*)d_in[48];
  const float* mlp_w1        = (const float*)d_in[49];
  const float* mlp_b1        = (const float*)d_in[50];
  const float* mlp_w2        = (const float*)d_in[51];
  const float* mlp_b2        = (const float*)d_in[52];

  // workspace arena (floats)
  float* ws = (float*)d_ws;
  size_t o = 0;
  float* x0   = ws + o; o += (size_t)Bq * Nn * 320;
  float* xcur = ws + o; o += (size_t)Bq * Nn * Hd;
  float* qf   = ws + o; o += (size_t)Bq * Nn * Hd;
  float* kf   = ws + o; o += (size_t)Bq * Nn * Hd;
  _Float16* qk16 = (_Float16*)(ws + o); o += (size_t)Bq * Nn * 512 / 2;   // MHA Q,K f16
  _Float16* vt16 = (_Float16*)(ws + o); o += (size_t)Bq * 256 * Nn / 2;   // head-transposed V f16
  float* sqb  = ws + o; o += (size_t)Bq * NHEADS * Nn;
  float* skb  = ws + o; o += (size_t)Bq * NHEADS * Nn;
  float* pb   = ws + o; o += (size_t)Bq * NHEADS * Nn * ETYPES;
  float* attn = ws + o; o += (size_t)Bq * Nn * Hd;
  unsigned char* mbytes = (unsigned char*)(ws + o); o += (size_t)Bq * Nn * Nn / 4;

  float* tok  = (float*)d_out;                       // graph_tokens [B,N,256]
  float* gr   = tok + (size_t)Bq * Nn * Hd;          // global_repr [B,N,256]
  float* gemb = gr  + (size_t)Bq * Nn * Hd;          // graph_embedding [B,256]

  const dim3 gg(Hd / 64, (Bq * Nn) / 128);

  // 1) pack adjacency + edge types -> byte mask
  prep_mask<<<(Bq * Nn * Nn) / 256, 256, 0, stream>>>(etm, adj1, adj2, adj3, mbytes);

  // 2) embed: relu(nf @ emb_w + b) into x0[:, :256] (ldc=320), plus type embed cols
  gemm_wmma<1,0,0,1><<<gg, 256, 0, stream>>>(
      node_features, NFD, emb_w, Hd, emb_b, nullptr, x0, 320, nullptr, 0, NFD);
  type_gather<<<(Bq * Nn * 64) / 256, 256, 0, stream>>>(node_types, type_emb, x0);

  // 3) GAT layers
  for (int L = 0; L < 3; ++L) {
    int bi = 9 + 12 * L;
    const float* wq  = (const float*)d_in[bi + 0];
    const float* bqv = (const float*)d_in[bi + 1];
    const float* wk  = (const float*)d_in[bi + 2];
    const float* bk  = (const float*)d_in[bi + 3];
    const float* wv  = (const float*)d_in[bi + 4];
    const float* bv  = (const float*)d_in[bi + 5];
    const float* attw= (const float*)d_in[bi + 6];
    const float* ete = (const float*)d_in[bi + 7];
    const float* wo  = (const float*)d_in[bi + 8];
    const float* bo  = (const float*)d_in[bi + 9];
    const float* lng = (const float*)d_in[bi + 10];
    const float* lnb = (const float*)d_in[bi + 11];
    const float* xin = L ? xcur : x0;
    int lda = L ? Hd : 320;
    gemm_wmma<1,0,0,0><<<gg, 256, 0, stream>>>(xin, lda, wq, Hd, bqv, nullptr,
                                               qf, Hd, nullptr, 0, lda);
    gemm_wmma<1,0,0,0><<<gg, 256, 0, stream>>>(xin, lda, wk, Hd, bk, nullptr,
                                               kf, Hd, nullptr, 0, lda);
    gemm_wmma<0,2,0,0><<<gg, 256, 0, stream>>>(xin, lda, wv, Hd, bv, nullptr,
                                               nullptr, Hd, vt16, 0, lda);
    gat_pre<<<(Bq * Nn * NHEADS * 12) / 256, 256, 0, stream>>>(qf, kf, attw, ete,
                                                               sqb, skb, pb);
    gat_attn<<<Bq * NHEADS * (Nn / 16) / 4, 128, 0, stream>>>(sqb, skb, pb, mbytes,
                                                              vt16, attn);
    if (L == 0)
      gemm_wmma<1,0,0,0><<<gg, 256, 0, stream>>>(attn, Hd, wo, Hd, bo, nullptr,
                                                 xcur, Hd, nullptr, 0, Hd);
    else
      gemm_wmma<1,0,1,0><<<gg, 256, 0, stream>>>(attn, Hd, wo, Hd, bo, xcur,
                                                 xcur, Hd, nullptr, 0, Hd);
    float* lnout = (L == 2) ? tok : xcur;
    layernorm_k<<<Bq * Nn, 256, 0, stream>>>(xcur, lng, lnb, lnout);
  }

  // 4) MHA: f16-only Q/K GEMM (ld=512), head-transposed V GEMM, flash attention,
  //    out-proj into global_repr
  gemm_wmma<0,1,0,0><<<dim3(512 / 64, (Bq * Nn) / 128), 256, 0, stream>>>(
      tok, Hd, mha_in_w, 768, mha_in_b, nullptr, nullptr, 0, qk16, 512, Hd);
  gemm_wmma<0,2,0,0><<<gg, 256, 0, stream>>>(
      tok, Hd, mha_in_w + 512, 768, mha_in_b + 512, nullptr, nullptr, 0, vt16, 0, Hd);
  float* mhatt = x0;  // reuse
  mha_attn<<<Bq * NHEADS * (Nn / 16) / 4, 128, 0, stream>>>(qk16, vt16, mhatt);
  gemm_wmma<1,0,0,0><<<gg, 256, 0, stream>>>(
      mhatt, Hd, mha_out_w, Hd, mha_out_b, nullptr, gr, Hd, nullptr, 0, Hd);

  // 5) mean-pool + MLP -> graph_embedding
  pool_mlp<<<Bq, 256, 0, stream>>>(gr, mlp_w1, mlp_b1, mlp_w2, mlp_b2, gemb);
}
